// MultiHeadAttention_37623913513495
// MI455X (gfx1250) — matrix operand order repaired and checked by reading, not compiled, not run on hardware
//
#include <hip/hip_runtime.h>
#include <hip/hip_bf16.h>

typedef __bf16 bf16_t;
typedef __attribute__((ext_vector_type(16))) __bf16 v16bf;
typedef __attribute__((ext_vector_type(8)))  float  v8f;

#define N_HEADC 16
#define HEADC   64
#define SEQC    1024
#define EMBC    1024
#define BATCHC  4

union FragAB { unsigned int u[8]; v16bf v; };

__device__ __forceinline__ v8f wmma_bf16(const FragAB& a, const FragAB& b, v8f c) {
  return __builtin_amdgcn_wmma_f32_16x16x32_bf16(false, a.v, false, b.v, (short)0, c,
                                                 false, false);
}

// 16-bit A 16x32 fragment, contiguous k order used here (same as B loads): lane = M (lane&15), K pairs across VGPRs,
// lane-half selects K half: k = 16*lh + 2*j + {0,1}.
__device__ __forceinline__ int a_kbase(int j, int lh) {
  return 2 * j + 16 * lh;
}

// Async DMA one 128-byte row (64 bf16) from global memory into LDS.
// ISA 10.x: the instruction offset applies to the LDS address as well as the global one,
// so the same offset immediate advances both addresses. Tracked by ASYNCcnt.
__device__ __forceinline__ void async_copy_row128(unsigned lds_off, const bf16_t* gsrc) {
  unsigned long long ga = (unsigned long long)gsrc;
  asm volatile(
      "global_load_async_to_lds_b128 %0, %1, off\n\t"
      "global_load_async_to_lds_b128 %0, %1, off offset:16\n\t"
      "global_load_async_to_lds_b128 %0, %1, off offset:32\n\t"
      "global_load_async_to_lds_b128 %0, %1, off offset:48\n\t"
      "global_load_async_to_lds_b128 %0, %1, off offset:64\n\t"
      "global_load_async_to_lds_b128 %0, %1, off offset:80\n\t"
      "global_load_async_to_lds_b128 %0, %1, off offset:96\n\t"
      "global_load_async_to_lds_b128 %0, %1, off offset:112"
      :: "v"(lds_off), "v"(ga) : "memory");
}

__device__ __forceinline__ void wait_async0() {
  asm volatile("s_wait_asynccnt 0x0" ::: "memory");
}

// ---------------------------------------------------------------------------
// fp32 -> bf16 elementwise
__global__ __launch_bounds__(256)
void cvt_bf16_kernel(const float* __restrict__ s, bf16_t* __restrict__ d, int n) {
  int i = blockIdx.x * 256 + threadIdx.x;
  if (i < n) d[i] = (bf16_t)s[i];
}

// fp32 [1024 k][1024 n] -> bf16 transposed [n][k] so B-fragment K-pairs are contiguous
__global__ __launch_bounds__(256)
void cvt_tr_kernel(const float* __restrict__ s, bf16_t* __restrict__ d) {
  int i = blockIdx.x * 256 + threadIdx.x;  // i < 1024*1024
  int k = i >> 10, n = i & 1023;
  d[(size_t)n * EMBC + k] = (bf16_t)s[i];
}

// ---------------------------------------------------------------------------
// Fused QKV projection + rotary. One wave -> 32 rows x 64 cols (one full head),
// so each B fragment feeds two WMMAs. grid = (4096/32, 16 heads, 3 {q,k,v})
__global__ __launch_bounds__(32)
void proj_rope_kernel(const bf16_t* __restrict__ xb,
                      const bf16_t* __restrict__ wqT,
                      const bf16_t* __restrict__ wkT,
                      const bf16_t* __restrict__ wvT,
                      bf16_t* __restrict__ Qb, bf16_t* __restrict__ Kb,
                      bf16_t* __restrict__ Vt) {
  const int lane = threadIdx.x & 31;
  const int ln = lane & 15, lh = lane >> 4;
  const int m0 = blockIdx.x * 32;
  const int h  = blockIdx.y;
  const int mode = blockIdx.z;
  const bf16_t* WT = (mode == 0) ? wqT : (mode == 1) ? wkT : wvT;

  v8f acc[2][4] = {};
  for (int kk = 0; kk < EMBC; kk += 32) {
    FragAB a[2];
#pragma unroll
    for (int mi = 0; mi < 2; ++mi)
#pragma unroll
      for (int j = 0; j < 8; ++j)
        a[mi].u[j] = *(const unsigned int*)
            &xb[(size_t)(m0 + 16 * mi + ln) * EMBC + kk + a_kbase(j, lh)];
#pragma unroll
    for (int f = 0; f < 4; ++f) {
      FragAB bfr;
      const int n = h * HEADC + 16 * f + ln;
#pragma unroll
      for (int j = 0; j < 8; ++j)
        bfr.u[j] = *(const unsigned int*)&WT[(size_t)n * EMBC + kk + 2 * j + 16 * lh];
      acc[0][f] = wmma_bf16(a[0], bfr, acc[0][f]);
      acc[1][f] = wmma_bf16(a[1], bfr, acc[1][f]);
    }
  }

#pragma unroll
  for (int mi = 0; mi < 2; ++mi) {
#pragma unroll
    for (int r = 0; r < 8; ++r) {
      const int m = 16 * mi + r + 8 * lh;
      const int grow = m0 + m;
      const int b = grow >> 10, t = grow & 1023;
      if (mode < 2) {
        // rotary: partner dim d +/- 32 lives in fragment f^2, same row/lane
        float outv[4];
#pragma unroll
        for (int f = 0; f < 4; ++f) {
          const int d = 16 * f + ln;
          const float vcur = acc[mi][f][r];
          const float vpar = acc[mi][f ^ 2][r];
          const int j32 = d & 31;
          // inv_freq = 10000^(-j32/32) = exp2(-j32 * log2(10000)/32)
          const float invf = exp2f(-0.41524101186f * (float)j32);
          float sn, cs;
          __sincosf((float)t * invf, &sn, &cs);
          const float rot = (d < 32) ? -vpar : vpar;
          outv[f] = vcur * cs + rot * sn;
        }
        bf16_t* dst = (mode == 0) ? Qb : Kb;
#pragma unroll
        for (int f = 0; f < 4; ++f) {
          const int d = 16 * f + ln;
          dst[(((size_t)b * N_HEADC + h) * SEQC + t) * HEADC + d] = (bf16_t)outv[f];
        }
      } else {
#pragma unroll
        for (int f = 0; f < 4; ++f) {
          const int d = 16 * f + ln;
          Vt[(((size_t)b * N_HEADC + h) * HEADC + d) * SEQC + t] = (bf16_t)acc[mi][f][r];
        }
      }
    }
  }
}

// ---------------------------------------------------------------------------
// Flash attention with rel-pos skew trick. One wave per (b, h, 16-row i-tile).
// K/rel tiles staged with global_load_async_to_lds_b128 (ASYNCcnt). grid=(64,16,4)
__global__ __launch_bounds__(32)
void attn_kernel(const bf16_t* __restrict__ Qb, const bf16_t* __restrict__ Kb,
                 const bf16_t* __restrict__ Vt, const bf16_t* __restrict__ relb,
                 bf16_t* __restrict__ Ob) {
  __shared__ bf16_t lds_k[32 * HEADC];     // K tile [j][d]
  __shared__ bf16_t lds_rel[48 * HEADC];   // rel rows [u][d]
  __shared__ float  lds_qr[16 * 48];       // Q . rel^T diagonals
  __shared__ bf16_t lds_p[16 * 32];        // P tile for layout change C->A

  const int lane = threadIdx.x & 31;
  const int ln = lane & 15, lh = lane >> 4;
  const int i0 = blockIdx.x * 16;
  const int h = blockIdx.y, b = blockIdx.z;
  const size_t bh = (size_t)b * N_HEADC + h;
  const float scale = 0.125f;  // 1/sqrt(64)

  // Q fragments: K=0..31 and K=32..63 (A-layout directly from global, u32 pairs)
  FragAB qf[2];
#pragma unroll
  for (int kf = 0; kf < 2; ++kf)
#pragma unroll
    for (int j = 0; j < 8; ++j)
      qf[kf].u[j] = *(const unsigned int*)
          &Qb[(bh * SEQC + i0 + ln) * HEADC + 32 * kf + a_kbase(j, lh)];

  float mrow[8], lrow[8];
  v8f o[4] = {};
#pragma unroll
  for (int r = 0; r < 8; ++r) { mrow[r] = -INFINITY; lrow[r] = 0.0f; }

  for (int j0 = 0; j0 < SEQC; j0 += 32) {
    __syncthreads();
    // stage K tile: one row (128B) per lane, async DMA global -> LDS
    async_copy_row128((unsigned)(size_t)&lds_k[lane * HEADC],
                      &Kb[(bh * SEQC + j0 + lane) * HEADC]);
    // stage rel rows: u = c - r + (T-1); 47 diagonals needed for this tile pair
    const int u0 = j0 - i0 + (SEQC - 1) - 15;
    {
      int u = u0 + lane;
      u = u < 0 ? 0 : (u > 2 * SEQC - 2 ? 2 * SEQC - 2 : u);
      async_copy_row128((unsigned)(size_t)&lds_rel[lane * HEADC],
                        &relb[(size_t)u * HEADC]);
      if (lane < 16) {
        int u2 = u0 + 32 + lane;
        u2 = u2 < 0 ? 0 : (u2 > 2 * SEQC - 2 ? 2 * SEQC - 2 : u2);
        async_copy_row128((unsigned)(size_t)&lds_rel[(32 + lane) * HEADC],
                          &relb[(size_t)u2 * HEADC]);
      }
    }
    if (j0 + 32 < SEQC)
      __builtin_prefetch(&Kb[(bh * SEQC + j0 + 32 + lane) * HEADC], 0, 1);
    wait_async0();
    __syncthreads();

    // QR = Q x rel^T over the 48-wide u window: 3 N-tiles x 2 K-steps
#pragma unroll
    for (int nt = 0; nt < 3; ++nt) {
      v8f qr = {};
#pragma unroll
      for (int kf = 0; kf < 2; ++kf) {
        FragAB bfr;
#pragma unroll
        for (int j = 0; j < 8; ++j)
          bfr.u[j] = *(const unsigned int*)
              &lds_rel[(16 * nt + ln) * HEADC + 32 * kf + 2 * j + 16 * lh];
        qr = wmma_bf16(qf[kf], bfr, qr);
      }
#pragma unroll
      for (int r = 0; r < 8; ++r)
        lds_qr[(r + 8 * lh) * 48 + 16 * nt + ln] = qr[r];
    }
    __syncthreads();

    // S = Q K^T * scale + gather(QR); two 16-col halves
    float pv[2][8];
    float rmax[8];
#pragma unroll
    for (int r = 0; r < 8; ++r) rmax[r] = -INFINITY;
#pragma unroll
    for (int nh = 0; nh < 2; ++nh) {
      v8f s = {};
#pragma unroll
      for (int kf = 0; kf < 2; ++kf) {
        FragAB bfr;
#pragma unroll
        for (int j = 0; j < 8; ++j)
          bfr.u[j] = *(const unsigned int*)
              &lds_k[(16 * nh + ln) * HEADC + 32 * kf + 2 * j + 16 * lh];
        s = wmma_bf16(qf[kf], bfr, s);
      }
#pragma unroll
      for (int r = 0; r < 8; ++r) {
        const int row = r + 8 * lh;
        const int c = 16 * nh + ln;
        const float val = s[r] * scale + lds_qr[row * 48 + (c - row + 15)];
        pv[nh][r] = val;
        rmax[r] = fmaxf(rmax[r], val);
      }
    }
    // row reductions across the 16-lane half (C-layout rows)
#pragma unroll
    for (int r = 0; r < 8; ++r)
#pragma unroll
      for (int off = 8; off >= 1; off >>= 1)
        rmax[r] = fmaxf(rmax[r], __shfl_xor(rmax[r], off, 32));

    float alpha[8];
#pragma unroll
    for (int r = 0; r < 8; ++r) {
      const float mnew = fmaxf(mrow[r], rmax[r]);
      alpha[r] = __expf(mrow[r] - mnew);
      mrow[r] = mnew;
      const float p0 = __expf(pv[0][r] - mnew);
      const float p1 = __expf(pv[1][r] - mnew);
      pv[0][r] = p0; pv[1][r] = p1;
      float rs = p0 + p1;
#pragma unroll
      for (int off = 8; off >= 1; off >>= 1)
        rs += __shfl_xor(rs, off, 32);
      lrow[r] = lrow[r] * alpha[r] + rs;
    }

    __syncthreads();
#pragma unroll
    for (int r = 0; r < 8; ++r) {
      const int row = r + 8 * lh;
      lds_p[row * 32 + ln]      = (bf16_t)pv[0][r];
      lds_p[row * 32 + 16 + ln] = (bf16_t)pv[1][r];
#pragma unroll
      for (int f = 0; f < 4; ++f) o[f][r] *= alpha[r];
    }
    __syncthreads();

    // O += P x V ; P reloaded in A-layout, V^T B-fragments straight from global
    FragAB pa;
#pragma unroll
    for (int j = 0; j < 8; ++j)
      pa.u[j] = *(const unsigned int*)&lds_p[ln * 32 + a_kbase(j, lh)];
#pragma unroll
    for (int f = 0; f < 4; ++f) {
      FragAB bv;
#pragma unroll
      for (int j = 0; j < 8; ++j)
        bv.u[j] = *(const unsigned int*)
            &Vt[(bh * HEADC + 16 * f + ln) * SEQC + j0 + 2 * j + 16 * lh];
      o[f] = wmma_bf16(pa, bv, o[f]);
    }
  }

#pragma unroll
  for (int r = 0; r < 8; ++r) {
    const int row = r + 8 * lh;
    const int t = i0 + row;
    const float inv = 1.0f / lrow[r];
#pragma unroll
    for (int f = 0; f < 4; ++f) {
      const int d = 16 * f + ln;
      Ob[((size_t)b * SEQC + t) * EMBC + h * HEADC + d] = (bf16_t)(o[f][r] * inv);
    }
  }
}

// ---------------------------------------------------------------------------
// out = Ob @ wo + bo ; fp32 output, 32x64 tile per wave. grid = (4096/32, 1024/64)
__global__ __launch_bounds__(32)
void out_gemm_kernel(const bf16_t* __restrict__ Ob, const bf16_t* __restrict__ woT,
                     const float* __restrict__ bo, float* __restrict__ out) {
  const int lane = threadIdx.x & 31;
  const int ln = lane & 15, lh = lane >> 4;
  const int m0 = blockIdx.x * 32;
  const int n0 = blockIdx.y * 64;
  v8f acc[2][4] = {};
  for (int kk = 0; kk < EMBC; kk += 32) {
    FragAB a[2];
#pragma unroll
    for (int mi = 0; mi < 2; ++mi)
#pragma unroll
      for (int j = 0; j < 8; ++j)
        a[mi].u[j] = *(const unsigned int*)
            &Ob[(size_t)(m0 + 16 * mi + ln) * EMBC + kk + a_kbase(j, lh)];
#pragma unroll
    for (int f = 0; f < 4; ++f) {
      FragAB bfr;
#pragma unroll
      for (int j = 0; j < 8; ++j)
        bfr.u[j] = *(const unsigned int*)
            &woT[(size_t)(n0 + 16 * f + ln) * EMBC + kk + 2 * j + 16 * lh];
      acc[0][f] = wmma_bf16(a[0], bfr, acc[0][f]);
      acc[1][f] = wmma_bf16(a[1], bfr, acc[1][f]);
    }
  }
#pragma unroll
  for (int mi = 0; mi < 2; ++mi)
#pragma unroll
    for (int r = 0; r < 8; ++r) {
      const int grow = m0 + 16 * mi + r + 8 * lh;
#pragma unroll
      for (int f = 0; f < 4; ++f) {
        const int n = n0 + 16 * f + ln;
        out[(size_t)grow * EMBC + n] = acc[mi][f][r] + bo[n];
      }
    }
}

// ---------------------------------------------------------------------------
extern "C" void kernel_launch(void* const* d_in, const int* in_sizes, int n_in,
                              void* d_out, int out_size, void* d_ws, size_t ws_size,
                              hipStream_t stream) {
  (void)in_sizes; (void)n_in; (void)out_size; (void)ws_size;
  const float* x   = (const float*)d_in[0];
  const float* wq  = (const float*)d_in[1];
  const float* wk  = (const float*)d_in[2];
  const float* wv  = (const float*)d_in[3];
  const float* wo  = (const float*)d_in[4];
  const float* bo  = (const float*)d_in[5];
  const float* rel = (const float*)d_in[6];
  float* out = (float*)d_out;

  char* ws = (char*)d_ws;
  size_t off = 0;
  auto alloc = [&](size_t bytes) -> void* {
    void* p = ws + off;
    off = (off + bytes + 255) & ~(size_t)255;
    return p;
  };
  const size_t NTOK = (size_t)BATCHC * SEQC;           // 4096
  bf16_t* xb   = (bf16_t*)alloc(NTOK * EMBC * sizeof(bf16_t));
  bf16_t* wqT  = (bf16_t*)alloc((size_t)EMBC * EMBC * sizeof(bf16_t));
  bf16_t* wkT  = (bf16_t*)alloc((size_t)EMBC * EMBC * sizeof(bf16_t));
  bf16_t* wvT  = (bf16_t*)alloc((size_t)EMBC * EMBC * sizeof(bf16_t));
  bf16_t* woT  = (bf16_t*)alloc((size_t)EMBC * EMBC * sizeof(bf16_t));
  bf16_t* relb = (bf16_t*)alloc((size_t)(2 * SEQC - 1) * HEADC * sizeof(bf16_t));
  bf16_t* Qb   = (bf16_t*)alloc(NTOK * EMBC * sizeof(bf16_t));
  bf16_t* Kb   = (bf16_t*)alloc(NTOK * EMBC * sizeof(bf16_t));
  bf16_t* Vt   = (bf16_t*)alloc(NTOK * EMBC * sizeof(bf16_t));
  bf16_t* Ob   = (bf16_t*)alloc(NTOK * EMBC * sizeof(bf16_t));

  const int nx = (int)(NTOK * EMBC);
  const int nrel = (2 * SEQC - 1) * HEADC;
  cvt_bf16_kernel<<<(nx + 255) / 256, 256, 0, stream>>>(x, xb, nx);
  cvt_tr_kernel<<<(EMBC * EMBC) / 256, 256, 0, stream>>>(wq, wqT);
  cvt_tr_kernel<<<(EMBC * EMBC) / 256, 256, 0, stream>>>(wk, wkT);
  cvt_tr_kernel<<<(EMBC * EMBC) / 256, 256, 0, stream>>>(wv, wvT);
  cvt_tr_kernel<<<(EMBC * EMBC) / 256, 256, 0, stream>>>(wo, woT);
  cvt_bf16_kernel<<<(nrel + 255) / 256, 256, 0, stream>>>(rel, relb, nrel);

  proj_rope_kernel<<<dim3(NTOK / 32, N_HEADC, 3), 32, 0, stream>>>(
      xb, wqT, wkT, wvT, Qb, Kb, Vt);

  attn_kernel<<<dim3(SEQC / 16, N_HEADC, BATCHC), 32, 0, stream>>>(
      Qb, Kb, Vt, relb, Ob);

  out_gemm_kernel<<<dim3(NTOK / 32, EMBC / 64), 32, 0, stream>>>(Ob, woT, bo, out);
}
